// GraphWeatherForecaster_54022098649844
// MI455X (gfx1250) — matrix operand order repaired and checked by reading, not compiled, not run on hardware
//
#include <hip/hip_runtime.h>
#include <hip/hip_bf16.h>
#include <stdint.h>

typedef _Float16 h16;
typedef __attribute__((ext_vector_type(16))) _Float16 v16h;
typedef __attribute__((ext_vector_type(8)))  _Float16 v8h;
typedef __attribute__((ext_vector_type(8)))  float    v8f;

#define NGN 65160
#define NMN 5882
#define EMN 35292
#define DDIM 256
#define FDIM 78

#define SA 40   // LDS row stride (halves) for A tile (64 rows x 32 k)
#define SB 40   // LDS row stride (halves) for B tile (128 cols x 32 k, col-major-of-W)

__device__ __forceinline__ float silu_f(float x) { return x / (1.0f + __expf(-x)); }

// ---------------------------------------------------------------------------
// Tiled WMMA GEMM: out = act(A[MxK] @ W[KxN] + bias)
// A : f16 row-major, lda = K (padded, multiple of 32)
// Wt: f16 TRANSPOSED weights [N x K], ldw = Kpad (multiple of 32); N padded to 128
// 128 threads = 4 waves; WG tile 64x128; wave tile 32x64 = 2x4 WMMA 16x16 tiles
// ---------------------------------------------------------------------------
template <int ACT>
__global__ __launch_bounds__(128) void gemm_wmma(
    const h16* __restrict__ A, int lda,
    const h16* __restrict__ Wt, int ldw,
    const float* __restrict__ bias,
    h16* __restrict__ out16, int ldo16,
    float* __restrict__ out32, int ldo32,
    int M, int N, int K)
{
    __shared__ __align__(16) h16 Asm[64 * SA];
    __shared__ __align__(16) h16 Bsm[128 * SB];

    const int tid  = threadIdx.x;
    const int lane = tid & 31;
    const int wv   = tid >> 5;
    const int wr   = wv >> 1;      // wave row within WG (0..1)
    const int wc   = wv & 1;       // wave col within WG (0..1)
    const int m    = lane & 15;
    const int hf   = lane >> 4;    // half-wave select
    const int blockM = blockIdx.x * 64;
    const int blockN = blockIdx.y * 128;

    const v8f vzero = {0.f, 0.f, 0.f, 0.f, 0.f, 0.f, 0.f, 0.f};
    v8f acc[2][4];
#pragma unroll
    for (int i = 0; i < 2; i++)
#pragma unroll
        for (int j = 0; j < 4; j++) acc[i][j] = vzero;

    for (int kt = 0; kt < K; kt += 32) {
        // --- cooperative load of A tile (64 rows x 32 halves), zero-fill OOB rows
#pragma unroll
        for (int it = 0; it < 2; it++) {
            int c = tid + 128 * it;          // 0..255 chunks of 8 halves
            int row = c >> 2;
            int kc  = (c & 3) * 8;
            uint4 val = {0u, 0u, 0u, 0u};
            int gr = blockM + row;
            if (gr < M) val = *(const uint4*)(A + (size_t)gr * lda + kt + kc);
            *(uint4*)(&Asm[row * SA + kc]) = val;
        }
        // --- cooperative load of B tile from transposed weights:
        //     Bsm[col][k] = Wt[blockN+col][kt+k]; fully coalesced, no scalar scatter
#pragma unroll
        for (int it = 0; it < 4; it++) {
            int c = tid + 128 * it;          // 0..511 chunks of 8 halves
            int col = c >> 2;
            int kc  = (c & 3) * 8;
            uint4 val = *(const uint4*)(Wt + (size_t)(blockN + col) * ldw + kt + kc);
            *(uint4*)(&Bsm[col * SB + kc]) = val;
        }
        // --- prefetch next k-tile while this one is staged
        if (kt + 32 < K) {
            __builtin_prefetch(Wt + (size_t)(blockN + tid) * ldw + kt + 32, 0, 0);
            if (tid < 64) {
                int gr = blockM + tid;
                if (gr < M) __builtin_prefetch(A + (size_t)gr * lda + kt + 32, 0, 0);
            }
        }
        __syncthreads();

        // --- build fragments per ISA 16-bit A(16x32)/B(32x16) lane layouts
        v16h af[2], bf[4];
#pragma unroll
        for (int i = 0; i < 2; i++) {
            int r = wr * 32 + i * 16 + m;
            v8h lo = *(const v8h*)(&Asm[r * SA + hf * 8]);       // K = hf*8 .. +7
            v8h hi = *(const v8h*)(&Asm[r * SA + 16 + hf * 8]);  // K = 16+hf*8 .. +7
            af[i] = __builtin_shufflevector(lo, hi, 0, 1, 2, 3, 4, 5, 6, 7,
                                            8, 9, 10, 11, 12, 13, 14, 15);
        }
#pragma unroll
        for (int j = 0; j < 4; j++) {
            int cl = wc * 64 + j * 16 + m;
            v8h lo = *(const v8h*)(&Bsm[cl * SB + hf * 8]);      // K = hf*8 .. +7
            v8h hi = *(const v8h*)(&Bsm[cl * SB + 16 + hf * 8]); // K = 16+hf*8 .. +7
            bf[j] = __builtin_shufflevector(lo, hi, 0, 1, 2, 3, 4, 5, 6, 7,
                                            8, 9, 10, 11, 12, 13, 14, 15);
        }
#pragma unroll
        for (int i = 0; i < 2; i++)
#pragma unroll
            for (int j = 0; j < 4; j++)
                acc[i][j] = __builtin_amdgcn_wmma_f32_16x16x32_f16(
                    false, af[i], false, bf[j], (short)0, acc[i][j], false, false);
        __syncthreads();
    }

    // --- epilogue: bias + optional SiLU; C/D layout: VGPR v -> M = hf*8+v, N = m
#pragma unroll
    for (int i = 0; i < 2; i++) {
#pragma unroll
        for (int j = 0; j < 4; j++) {
#pragma unroll
            for (int v = 0; v < 8; v++) {
                int row = blockM + wr * 32 + i * 16 + hf * 8 + v;
                int col = blockN + wc * 64 + j * 16 + m;
                if (row < M && col < N) {
                    float f = acc[i][j][v] + bias[col];
                    if (ACT) f = silu_f(f);
                    if (out16) out16[(size_t)row * ldo16 + col] = (h16)f;
                    if (out32) out32[(size_t)row * ldo32 + col] = f;
                }
            }
        }
    }
}

// ---------------------------------------------------------------------------
// LayerNorm(h)*g + be  (+ optional residual), row length 256, 1 block per row
// ---------------------------------------------------------------------------
__global__ __launch_bounds__(256) void ln_res_kernel(
    const float* __restrict__ h, const float* __restrict__ res,
    const float* __restrict__ g, const float* __restrict__ be,
    float* __restrict__ out, int M)
{
    __shared__ float sred[256];
    const int row = blockIdx.x;
    const int t = threadIdx.x;
    const size_t idx = (size_t)row * 256 + t;
    float v = h[idx];
    sred[t] = v;
    __syncthreads();
    for (int s = 128; s > 0; s >>= 1) {
        if (t < s) sred[t] += sred[t + s];
        __syncthreads();
    }
    float mu = sred[0] * (1.0f / 256.0f);
    __syncthreads();
    float d = v - mu;
    sred[t] = d * d;
    __syncthreads();
    for (int s = 128; s > 0; s >>= 1) {
        if (t < s) sred[t] += sred[t + s];
        __syncthreads();
    }
    float var = sred[0] * (1.0f / 256.0f);
    float y = d * rsqrtf(var + 1e-5f) * g[t] + be[t];
    out[idx] = (res ? res[idx] : 0.0f) + y;
}

// concat [e | x_src[src] | x_dst[dst]] -> f16 [E x 768], one block per edge
__global__ __launch_bounds__(256) void edge_concat_kernel(
    const float* __restrict__ e, const float* __restrict__ xs,
    const float* __restrict__ xd, const int* __restrict__ src,
    const int* __restrict__ dst, h16* __restrict__ out, int E)
{
    const int i = blockIdx.x;
    const int t = threadIdx.x;
    const int s = src[i];
    const int d = dst[i];
    out[(size_t)i * 768 + t]       = (h16)e[(size_t)i * 256 + t];
    out[(size_t)i * 768 + 256 + t] = (h16)xs[(size_t)s * 256 + t];
    out[(size_t)i * 768 + 512 + t] = (h16)xd[(size_t)d * 256 + t];
}

// concat [a | b] -> f16 [M x 512], one block per row
__global__ __launch_bounds__(256) void concat2_kernel(
    const float* __restrict__ a, const float* __restrict__ b,
    h16* __restrict__ out, int M)
{
    const int i = blockIdx.x;
    const int t = threadIdx.x;
    out[(size_t)i * 512 + t]       = (h16)a[(size_t)i * 256 + t];
    out[(size_t)i * 512 + 256 + t] = (h16)b[(size_t)i * 256 + t];
}

// agg[dst[i]] += e[i]  (256 feats per edge)
__global__ __launch_bounds__(256) void scatter_add_kernel(
    const float* __restrict__ e, const int* __restrict__ dst,
    float* __restrict__ agg, int E)
{
    const int i = blockIdx.x;
    const int t = threadIdx.x;
    atomicAdd(&agg[(size_t)dst[i] * 256 + t], e[(size_t)i * 256 + t]);
}

__global__ __launch_bounds__(256) void zero_kernel(float* __restrict__ p, size_t n)
{
    size_t i = (size_t)blockIdx.x * 256 + threadIdx.x;
    if (i < n) p[i] = 0.0f;
}

// f32 [M x Kr] -> f16 [M x Kp], zero padded
__global__ __launch_bounds__(256) void stage_pad_kernel(
    const float* __restrict__ src, h16* __restrict__ dst, int M, int Kr, int Kp)
{
    size_t i = (size_t)blockIdx.x * 256 + threadIdx.x;
    if (i >= (size_t)M * Kp) return;
    int r = (int)(i / Kp);
    int k = (int)(i % Kp);
    dst[i] = (k < Kr) ? (h16)src[(size_t)r * Kr + k] : (h16)0.0f;
}

// weight f32 [Kr x Nr] -> TRANSPOSED f16 [Np x Kp], zero padded
__global__ __launch_bounds__(256) void cvt_w_kernel(
    const float* __restrict__ src, h16* __restrict__ dst,
    int Kr, int Nr, int Kp, int Np)
{
    size_t i = (size_t)blockIdx.x * 256 + threadIdx.x;
    if (i >= (size_t)Kp * Np) return;
    int n = (int)(i / Kp);
    int k = (int)(i % Kp);
    dst[i] = (k < Kr && n < Nr) ? (h16)src[(size_t)k * Nr + n] : (h16)0.0f;
}

// ---------------------------------------------------------------------------
// Host-side orchestration
// ---------------------------------------------------------------------------
struct LinW { const h16* W; const float* b; int Kp; int Np; };
struct MLP3 { LinW l0, l1, l2; const float* g; const float* be; };

extern "C" void kernel_launch(void* const* d_in, const int* in_sizes, int n_in,
                              void* d_out, int out_size, void* d_ws, size_t ws_size,
                              hipStream_t stream)
{
    (void)in_sizes; (void)n_in; (void)out_size; (void)ws_size;
    char* ws = (char*)d_ws;
    size_t off = 0;
    auto alloc = [&](size_t b) -> size_t {
        size_t o = off;
        off += (b + 255) & ~(size_t)255;
        return o;
    };
    const size_t off_cat = alloc((size_t)NGN * 768 * 2);  // f16 concat / staging
    const size_t off_ha  = alloc((size_t)NGN * 256 * 2);  // f16 hidden 1
    const size_t off_hb  = alloc((size_t)NGN * 256 * 2);  // f16 hidden 2
    const size_t off_xg  = alloc((size_t)NGN * 256 * 4);  // x_grid
    const size_t off_xm  = alloc((size_t)NMN * 256 * 4);  // x_mesh
    const size_t off_e   = alloc((size_t)NGN * 256 * 4);  // edge state (g2m/mesh/m2g)
    const size_t off_t   = alloc((size_t)NGN * 256 * 4);  // pre-LN MLP output
    const size_t off_agg = alloc((size_t)NGN * 256 * 4);  // scatter target / x_out
    size_t woff = off;                                    // f16 weights region

    auto F32 = [&](int i) -> const float* { return (const float*)d_in[i]; };
    auto I32 = [&](int i) -> const int* { return (const int*)d_in[i]; };

    auto cvt = [&](const float* src, int Kr, int Nr, int Kp, int Np) -> const h16* {
        h16* dst = (h16*)(ws + woff);
        woff += (((size_t)Kp * Np * 2) + 255) & ~(size_t)255;
        size_t n = (size_t)Kp * Np;
        cvt_w_kernel<<<(unsigned)((n + 255) / 256), 256, 0, stream>>>(src, dst, Kr, Nr, Kp, Np);
        return dst;
    };
    auto mk = [&](const float* W0, const float* b0, const float* W1, const float* b1,
                  const float* W2, const float* b2, const float* g, const float* be,
                  int K0r, int K0p, int dh, int Nr, int Np) -> MLP3 {
        MLP3 m;
        m.l0 = LinW{cvt(W0, K0r, dh, K0p, dh), b0, K0p, dh};
        m.l1 = LinW{cvt(W1, dh, dh, dh, dh), b1, dh, dh};
        m.l2 = LinW{cvt(W2, dh, Nr, dh, Np), b2, dh, Np};
        m.g = g;
        m.be = be;
        return m;
    };
    auto mk8 = [&](int base, int K0r, int K0p, int dh, int Nr, int Np, bool ln) -> MLP3 {
        return mk(F32(base), F32(base + 1), F32(base + 2), F32(base + 3),
                  F32(base + 4), F32(base + 5),
                  ln ? F32(base + 6) : nullptr, ln ? F32(base + 7) : nullptr,
                  K0r, K0p, dh, Nr, Np);
    };

    // input index map (pytree-flattened setup_inputs order)
    MLP3 enc_node = mk8(10, 78, 96, 256, 256, 256, true);
    MLP3 enc_edge = mk8(18, 3, 32, 256, 256, 256, true);
    MLP3 enc_be   = mk8(26, 768, 768, 256, 256, 256, true);
    MLP3 enc_bn   = mk8(34, 512, 512, 256, 256, 256, true);
    MLP3 pe[9], pn[9];
    {
        const float *W0 = F32(42), *b0 = F32(43), *W1 = F32(44), *b1 = F32(45),
                    *W2 = F32(46), *b2 = F32(47), *g = F32(48), *be = F32(49);
        for (int k = 0; k < 9; k++)
            pe[k] = mk(W0 + (size_t)k * 768 * 256, b0 + k * 256,
                       W1 + (size_t)k * 256 * 256, b1 + k * 256,
                       W2 + (size_t)k * 256 * 256, b2 + k * 256,
                       g + k * 256, be + k * 256, 768, 768, 256, 256, 256);
    }
    {
        const float *W0 = F32(50), *b0 = F32(51), *W1 = F32(52), *b1 = F32(53),
                    *W2 = F32(54), *b2 = F32(55), *g = F32(56), *be = F32(57);
        for (int k = 0; k < 9; k++)
            pn[k] = mk(W0 + (size_t)k * 512 * 256, b0 + k * 256,
                       W1 + (size_t)k * 256 * 256, b1 + k * 256,
                       W2 + (size_t)k * 256 * 256, b2 + k * 256,
                       g + k * 256, be + k * 256, 512, 512, 256, 256, 256);
    }
    MLP3 dec_edge = mk8(58, 3, 32, 256, 256, 256, true);
    MLP3 dec_be   = mk8(66, 768, 768, 256, 256, 256, true);
    MLP3 dec_bn   = mk8(74, 512, 512, 256, 256, 256, true);
    MLP3 dec_mlp  = mk8(82, 512, 512, 128, 78, 128, false);

    h16* CAT   = (h16*)(ws + off_cat);
    h16* HA    = (h16*)(ws + off_ha);
    h16* HB    = (h16*)(ws + off_hb);
    float* XG  = (float*)(ws + off_xg);
    float* XM  = (float*)(ws + off_xm);
    float* EB  = (float*)(ws + off_e);
    float* T32 = (float*)(ws + off_t);
    float* AGG = (float*)(ws + off_agg);

    auto gemm = [&](const h16* A, int lda, const LinW& L, h16* o16, int ldo16,
                    float* o32, int ldo32, int M, int Nreal, bool act) {
        dim3 g((unsigned)((M + 63) / 64), (unsigned)((Nreal + 127) / 128));
        if (act)
            gemm_wmma<1><<<g, 128, 0, stream>>>(A, lda, L.W, L.Kp, L.b,
                                                o16, ldo16, o32, ldo32, M, Nreal, L.Kp);
        else
            gemm_wmma<0><<<g, 128, 0, stream>>>(A, lda, L.W, L.Kp, L.b,
                                                o16, ldo16, o32, ldo32, M, Nreal, L.Kp);
    };
    auto run_mlp = [&](const h16* A, int lda, const MLP3& m, int M, int NoutReal,
                       const float* res, float* out) {
        int dh = m.l0.Np;
        gemm(A, lda, m.l0, HA, dh, nullptr, 0, M, dh, true);
        gemm(HA, dh, m.l1, HB, dh, nullptr, 0, M, dh, true);
        if (m.g) {
            gemm(HB, dh, m.l2, nullptr, 0, T32, 256, M, 256, false);
            ln_res_kernel<<<(unsigned)M, 256, 0, stream>>>(T32, res, m.g, m.be, out, M);
        } else {
            gemm(HB, dh, m.l2, nullptr, 0, out, NoutReal, M, NoutReal, false);
        }
    };
    auto block = [&](const float* xsrc, const float* xdst, float* e,
                     const int* src, const int* dst, int E, int Nd,
                     const MLP3& mep, const MLP3& mnp, float* xout) {
        edge_concat_kernel<<<(unsigned)E, 256, 0, stream>>>(e, xsrc, xdst, src, dst, CAT, E);
        run_mlp(CAT, 768, mep, E, 256, e, e);  // e += LN(mlp(concat))
        zero_kernel<<<(unsigned)(((size_t)Nd * 256 + 255) / 256), 256, 0, stream>>>(
            AGG, (size_t)Nd * 256);
        scatter_add_kernel<<<(unsigned)E, 256, 0, stream>>>(e, dst, AGG, E);
        concat2_kernel<<<(unsigned)Nd, 256, 0, stream>>>(xdst, AGG, CAT, Nd);
        run_mlp(CAT, 512, mnp, Nd, 256, xdst, xout);  // x = xdst + LN(mlp(concat))
    };

    // ------------------- forward pass -------------------
    // encoder node MLP: x_grid = mlp(features)
    stage_pad_kernel<<<(unsigned)(((size_t)NGN * 96 + 255) / 256), 256, 0, stream>>>(
        F32(0), CAT, NGN, 78, 96);
    run_mlp(CAT, 96, enc_node, NGN, 256, nullptr, XG);

    // e_g2m = mlp(g2m_ef)
    stage_pad_kernel<<<(unsigned)(((size_t)NGN * 32 + 255) / 256), 256, 0, stream>>>(
        F32(3), CAT, NGN, 3, 32);
    run_mlp(CAT, 32, enc_edge, NGN, 256, nullptr, EB);

    // x_mesh = 0; encoder block (grid -> mesh)
    zero_kernel<<<(unsigned)(((size_t)NMN * 256 + 255) / 256), 256, 0, stream>>>(
        XM, (size_t)NMN * 256);
    block(XG, XM, EB, I32(1), I32(2), NGN, NMN, enc_be, enc_bn, XM);

    // e_mesh = mlp(mesh_ef)
    stage_pad_kernel<<<(unsigned)(((size_t)EMN * 32 + 255) / 256), 256, 0, stream>>>(
        F32(6), CAT, EMN, 3, 32);
    run_mlp(CAT, 32, enc_edge, EMN, 256, nullptr, EB);

    // 9 processor blocks (mesh -> mesh)
    for (int k = 0; k < 9; k++)
        block(XM, XM, EB, I32(4), I32(5), EMN, NMN, pe[k], pn[k], XM);

    // e_m2g = mlp(m2g_ef) with dec_edge_mlp
    stage_pad_kernel<<<(unsigned)(((size_t)NGN * 32 + 255) / 256), 256, 0, stream>>>(
        F32(9), CAT, NGN, 3, 32);
    run_mlp(CAT, 32, dec_edge, NGN, 256, nullptr, EB);

    // decoder block (mesh -> grid); x_out goes to AGG (x_grid preserved)
    block(XM, XG, EB, I32(7), I32(8), NGN, NGN, dec_be, dec_bn, AGG);

    // out = mlp(concat([x_out, x_grid]))  (no LN)
    concat2_kernel<<<(unsigned)NGN, 256, 0, stream>>>(AGG, XG, CAT, NGN);
    run_mlp(CAT, 512, dec_mlp, NGN, FDIM, nullptr, (float*)d_out);
}
